// spike_encoding_44100724195637
// MI455X (gfx1250) — matrix operand order repaired and checked by reading, not compiled, not run on hardware
//
#include <hip/hip_runtime.h>

typedef __attribute__((ext_vector_type(16))) _Float16 v16h;
typedef __attribute__((ext_vector_type(8)))  float    v8f;

#define C_IN   64
#define C_OUT  64
#define KS     8
#define NLEN   4096
#define TB     128          // T*B
#define KTOT   512          // C_IN*KS
#define NT     256          // positions per block tile
#define XS     264          // NT + 7 halo + 1 pad (lds row stride, f16 elems)
#define CNT_PER_CH 524288.0f // TB*NLEN

// dynamic LDS partition (bytes)
#define SM_W    0                       // 64 KiB f16 weights [co][512]
#define SM_XS   65536                   // 33792 B f16 im2col slab
#define SM_SUM  (65536 + 33792)         // 256 B
#define SM_SQ   (SM_SUM + 256)          // 256 B
#define SM_TOTAL (SM_SQ + 256)          // 99840 B  (< 320 KB/WGP)

// ---------------- kernel 0: weights f32 -> f16, zero BN accumulators ----------
__global__ void k_prep(const float* __restrict__ w, _Float16* __restrict__ wh,
                       float* __restrict__ stats) {
    int gid = blockIdx.x * blockDim.x + threadIdx.x;
    if (gid < C_OUT * KTOT) wh[gid] = (_Float16)w[gid];   // [co][ci*8+kk], same flat order
    if (gid < 256) stats[gid] = 0.0f;                     // sum, sq, scale, shift slots
}

// ---------------- kernel 1: fused im2col + WMMA GEMM + BN partial stats -------
__global__ __launch_bounds__(256)
void k_conv_wmma(const float* __restrict__ x, const _Float16* __restrict__ wh,
                 float* __restrict__ out, float* __restrict__ stats) {
    extern __shared__ char smem[];
    _Float16* wlds  = (_Float16*)(smem + SM_W);
    _Float16* xs    = (_Float16*)(smem + SM_XS);
    float*    ls_sum = (float*)(smem + SM_SUM);
    float*    ls_sq  = (float*)(smem + SM_SQ);

    const int tb = blockIdx.y;
    const int n0 = blockIdx.x * NT;

    if (threadIdx.x < C_OUT) { ls_sum[threadIdx.x] = 0.f; ls_sq[threadIdx.x] = 0.f; }

    // ---- async DMA: 64 KiB f16 weights -> LDS (16 B/lane x 16 issues), on ASYNCcnt
    {
        const unsigned lbase = (unsigned)(size_t)wlds;
        const unsigned long long gbase = (unsigned long long)(size_t)wh;
        #pragma unroll
        for (int i = 0; i < 16; ++i) {
            const int byte_off = (threadIdx.x + i * 256) * 16;
            unsigned la = lbase + byte_off;
            unsigned long long ga = gbase + byte_off;
            asm volatile("global_load_async_to_lds_b128 %0, %1, off"
                         :: "v"(la), "v"(ga) : "memory");
        }
    }

    // ---- overlapped: stage x slab (f32 -> f16), left pad 3 / right pad 4, zero OOB
    for (int i = threadIdx.x; i < C_IN * XS; i += 256) {
        int ci = i / XS;
        int j  = i - ci * XS;
        int n  = n0 + j - 3;
        float v = (n >= 0 && n < NLEN) ? x[(tb * C_IN + ci) * NLEN + n] : 0.0f;
        xs[i] = (_Float16)v;
    }

    asm volatile("s_wait_asynccnt 0" ::: "memory");
    __syncthreads();

    const int lane = threadIdx.x & 31;
    const int wave = threadIdx.x >> 5;
    const int row  = lane & 15;
    const int hi   = lane >> 4;
    const int m0   = wave * 32;          // 32 positions per wave
    const int pos0 = m0 + row;

    v8f acc0[4] = {};                    // mi=0, ni=0..3
    v8f acc1[4] = {};                    // mi=1

    #pragma unroll
    for (int kb = 0; kb < 16; ++kb) {
        // A fragments: this code's contiguous K order (lane row = M; K=e+16*hi), same as B
        const int ciA = kb * 4 + 2 * hi;
        const _Float16* xa = &xs[ciA * XS + pos0];
        v16h a0, a1;
        #pragma unroll
        for (int e = 0; e < 8; ++e) {
            a0[e]     = xa[e];
            a0[e + 8] = xa[XS + e];
            a1[e]     = xa[16 + e];
            a1[e + 8] = xa[XS + 16 + e];
        }
        // B fragments from LDS: lane = N (co); lanes<16 K=0..15, lanes>=16 K=16..31
        const int k0 = kb * 32 + 16 * hi;
        #pragma unroll
        for (int ni = 0; ni < 4; ++ni) {
            int co = ni * 16 + row;
            v16h b = *(const v16h*)(wlds + co * KTOT + k0);
            acc0[ni] = __builtin_amdgcn_wmma_f32_16x16x32_f16(
                false, a0, false, b, (short)0, acc0[ni], false, false);
            acc1[ni] = __builtin_amdgcn_wmma_f32_16x16x32_f16(
                false, a1, false, b, (short)0, acc1[ni], false, false);
        }
    }

    // store conv output + accumulate BN partials
    #pragma unroll
    for (int ni = 0; ni < 4; ++ni) {
        const int co = ni * 16 + row;
        float s = 0.f, q = 0.f;
        #pragma unroll
        for (int mi = 0; mi < 2; ++mi) {
            v8f A = mi ? acc1[ni] : acc0[ni];
            int pos = n0 + m0 + mi * 16 + 8 * hi;      // element r -> pos + r
            float4* dst = (float4*)(out + (size_t)(tb * C_OUT + co) * NLEN + pos);
            float4 lo = {A[0], A[1], A[2], A[3]};
            float4 hv = {A[4], A[5], A[6], A[7]};
            dst[0] = lo;
            dst[1] = hv;
            #pragma unroll
            for (int r = 0; r < 8; ++r) { s += A[r]; q += A[r] * A[r]; }
        }
        unsafeAtomicAdd(&ls_sum[co], s);
        unsafeAtomicAdd(&ls_sq[co], q);
    }
    __syncthreads();
    if (threadIdx.x < C_OUT) {
        unsafeAtomicAdd(&stats[threadIdx.x], ls_sum[threadIdx.x]);
        unsafeAtomicAdd(&stats[C_OUT + threadIdx.x], ls_sq[threadIdx.x]);
    }
}

// ---------------- kernel 2: finalize BN -> per-channel scale/shift ------------
__global__ void k_bn_finalize(float* __restrict__ stats,
                              const float* __restrict__ gamma,
                              const float* __restrict__ beta) {
    int c = threadIdx.x;
    if (c < C_OUT) {
        float mean = stats[c] / CNT_PER_CH;
        float var  = stats[C_OUT + c] / CNT_PER_CH - mean * mean;
        float sc   = gamma[c] * rsqrtf(var + 1e-5f);
        stats[128 + c] = sc;                  // scale
        stats[192 + c] = beta[c] - mean * sc; // shift
    }
}

// ---------------- kernel 3: BN affine + 4-step LIF scan, in place -------------
__global__ __launch_bounds__(256)
void k_lif(float* __restrict__ out, const float* __restrict__ stats) {
    const int idx = blockIdx.x * blockDim.x + threadIdx.x;   // float4 index
    const int c = ((idx * 4) >> 12) & (C_OUT - 1);           // N=4096 floats per channel row
    const float sc = stats[128 + c];
    const float sh = stats[192 + c];
    float4* o4 = (float4*)out;
    const size_t tstep = (size_t)(TB / 4) * C_OUT * NLEN / 4; // B*C*N float4s = 2097152
    float v0 = 0.f, v1 = 0.f, v2 = 0.f, v3 = 0.f;
    #pragma unroll
    for (int t = 0; t < 4; ++t) {
        float4 y = o4[t * tstep + idx];
        float a0 = 0.5f * (v0 + (y.x * sc + sh));
        float a1 = 0.5f * (v1 + (y.y * sc + sh));
        float a2 = 0.5f * (v2 + (y.z * sc + sh));
        float a3 = 0.5f * (v3 + (y.w * sc + sh));
        float4 s;
        s.x = (a0 >= 1.0f) ? 1.0f : 0.0f;
        s.y = (a1 >= 1.0f) ? 1.0f : 0.0f;
        s.z = (a2 >= 1.0f) ? 1.0f : 0.0f;
        s.w = (a3 >= 1.0f) ? 1.0f : 0.0f;
        v0 = (a0 >= 1.0f) ? 0.0f : a0;
        v1 = (a1 >= 1.0f) ? 0.0f : a1;
        v2 = (a2 >= 1.0f) ? 0.0f : a2;
        v3 = (a3 >= 1.0f) ? 0.0f : a3;
        o4[t * tstep + idx] = s;
    }
}

extern "C" void kernel_launch(void* const* d_in, const int* in_sizes, int n_in,
                              void* d_out, int out_size, void* d_ws, size_t ws_size,
                              hipStream_t stream) {
    const float* x     = (const float*)d_in[0];
    const float* w     = (const float*)d_in[1];
    // d_in[2] = conv bias: cancels exactly under training-mode BN -> unused
    const float* gamma = (const float*)d_in[3];
    const float* beta  = (const float*)d_in[4];
    float* out = (float*)d_out;

    _Float16* wh = (_Float16*)d_ws;                       // 64 KiB f16 weights
    float* stats = (float*)((char*)d_ws + 65536);         // [sum64, sq64, scale64, shift64]

    k_prep<<<128, 256, 0, stream>>>(w, wh, stats);
    k_conv_wmma<<<dim3(NLEN / NT, TB), 256, SM_TOTAL, stream>>>(x, wh, out, stats);
    k_bn_finalize<<<1, 64, 0, stream>>>(stats, gamma, beta);
    k_lif<<<(TB / 4) * C_OUT * NLEN / 4 / 256, 256, 0, stream>>>(out, stats);
}
